// NCDEModel_45449343926625
// MI455X (gfx1250) — matrix operand order repaired and checked by reading, not compiled, not run on hardware
//
#include <hip/hip_runtime.h>
#include <hip/hip_bf16.h>
#include <cstdint>

// ---------------------------------------------------------------------------
// NCDE forward for MI455X (gfx1250, wave32, WMMA bf16)
//   B=512, L=64, D=16, H=256, V=512, STEPS=64
// One workgroup owns 16 batch rows and runs all 64 Euler steps.
// Weights converted to bf16 once per launch into d_ws (live in 192MB L2).
// Round 2 -> 3: branch-free fast tanh (v_exp_f32 + v_rcp_f32), phase C
// processes 2 N-tiles per A-fragment load (dual accumulators).
// ---------------------------------------------------------------------------

typedef __attribute__((ext_vector_type(16))) __bf16 v16bf;
typedef __attribute__((ext_vector_type(8)))  float  v8f;

#define Hn 256
#define Vd 512
#define STEPS 64
#define TPB 256

struct __align__(16) U4 { uint32_t x, y, z, w; };

__device__ __forceinline__ uint16_t f2bf(float x) {
    uint32_t u = __float_as_uint(x);
    uint32_t r = (u + 0x7FFFu + ((u >> 16) & 1u)) >> 16;
    return (uint16_t)r;
}
__device__ __forceinline__ float bf2f(uint16_t u) {
    return __uint_as_float(((uint32_t)u) << 16);
}

// Branch-free tanh: 1 - 2/(e^{2x}+1); exp2-based, saturates to +/-1.
__device__ __forceinline__ float fast_tanh(float x) {
    float e = __builtin_amdgcn_exp2f(x * 2.8853900817779268f); // 2*log2(e)
    return 1.0f - 2.0f * __builtin_amdgcn_rcpf(e + 1.0f);
}

// A-matrix (16x32, MxK) fragment: lane l -> row M=l&15, g=l>>4.
// halfwords j -> K = 16g + j (contiguous)
__device__ __forceinline__ v16bf load_a_frag(const uint16_t* p, int stride) {
    int l = threadIdx.x & 31;
    int r = l & 15, g = l >> 4;
    union { U4 u[2]; v16bf v; } x;
    x.u[0] = *(const U4*)(p + r * stride + 16 * g);
    x.u[1] = *(const U4*)(p + r * stride + 16 * g + 8);
    return x.v;
}

// B-matrix (32x16, KxN) fragment from row-major W[out][in]:
// lane l -> column N=l&15, g=l>>4; halfwords j -> K = 16g + j (contiguous)
__device__ __forceinline__ v16bf load_b_frag(const uint16_t* p, int stride) {
    int l = threadIdx.x & 31;
    int n = l & 15, g = l >> 4;
    const U4* q = (const U4*)(p + n * stride + 16 * g);
    union { U4 u[2]; v16bf v; } x;
    x.u[0] = q[0];
    x.u[1] = q[1];
    return x.v;
}

// In-place layernorm over rows of a 16 x 512 bf16 LDS tile.
__device__ __forceinline__ void layernorm_inplace(
    uint16_t* z, float* redS, float* redQ, float* mu, float* rs,
    const float* __restrict__ gamma, const float* __restrict__ beta, int tid)
{
    {   // partial sums: 16 threads per row, 32 strided elements each
        int r = tid >> 4, p = tid & 15;
        float s = 0.f, q = 0.f;
        #pragma unroll 8
        for (int j = 0; j < Vd / 16; ++j) {
            float x = bf2f(z[r * Vd + p + 16 * j]);
            s += x; q += x * x;
        }
        redS[tid] = s; redQ[tid] = q;
    }
    __syncthreads();
    if (tid < 16) {
        float s = 0.f, q = 0.f;
        #pragma unroll
        for (int p = 0; p < 16; ++p) { s += redS[tid * 16 + p]; q += redQ[tid * 16 + p]; }
        float m = s * (1.0f / Vd);
        float v = q * (1.0f / Vd) - m * m;
        mu[tid] = m; rs[tid] = rsqrtf(v + 1e-5f);
    }
    __syncthreads();
    for (int e = tid; e < 16 * Vd; e += TPB) {
        int r = e >> 9, c = e & (Vd - 1);
        float x = (bf2f(z[e]) - mu[r]) * rs[r] * gamma[c] + beta[c];
        z[e] = f2bf(x);
    }
    __syncthreads();
}

__global__ void __launch_bounds__(TPB, 1)
ncde_kernel(const float* __restrict__ coeffs,
            const float* __restrict__ W0g, const float* __restrict__ b0g,
            const float* __restrict__ b1g, const float* __restrict__ g1g, const float* __restrict__ be1g,
            const float* __restrict__ b2g, const float* __restrict__ g2g, const float* __restrict__ be2g,
            const float* __restrict__ b3g,
            const uint16_t* __restrict__ W1b, const uint16_t* __restrict__ W2b,
            const uint16_t* __restrict__ W3b,
            float* __restrict__ out)
{
    __shared__ __align__(16) float    h32[16 * Hn];   // persistent state (fp32) 16KB
    __shared__ __align__(16) uint16_t hA [16 * Hn];   // h as bf16 A-matrix       8KB
    __shared__ __align__(16) uint16_t zA [16 * Vd];   // layer-1 activations     16KB
    __shared__ __align__(16) uint16_t zB [16 * Vd];   // layer-2 activations     16KB
    __shared__ __align__(16) float    dXs[16 * 16];   // spline derivative        1KB
    __shared__ float redS[256], redQ[256], mu[16], rs[16];

    const int tid  = threadIdx.x;
    const int lane = tid & 31;
    const int wave = tid >> 5;
    const int g    = lane >> 4;
    const int ln   = lane & 15;
    const int bb   = blockIdx.x * 16;        // batch base
    const float dtf = 63.0f / 64.0f;

    // ---- h0 = tanh(X0 @ W0^T + b0); thread t computes column t for 16 rows
    {
        float w0r[16];
        #pragma unroll
        for (int d = 0; d < 16; ++d) w0r[d] = W0g[tid * 16 + d];
        float bias = b0g[tid];
        for (int m = 0; m < 16; ++m) {
            const float* x0 = coeffs + (size_t)(bb + m) * (63 * 4 * 16);
            float s = bias;
            #pragma unroll
            for (int d = 0; d < 16; ++d) s += x0[d] * w0r[d];
            h32[m * Hn + tid] = fast_tanh(s);
        }
    }
    __syncthreads();

    for (int step = 0; step < STEPS; ++step) {
        // ---- stage 1: spline derivative dX and bf16 copy of h
        {
            float tc = step * dtf;
            int idx = (int)tc; if (idx > 62) idx = 62;
            float frac = tc - (float)idx;
            int m = tid >> 4, d = tid & 15;
            const float* cp = coeffs + (((size_t)(bb + m) * 63 + idx) * 4) * 16 + d;
            dXs[tid] = cp[16] + 2.0f * frac * cp[32] + 3.0f * frac * frac * cp[48];
            for (int e = tid; e < 16 * Hn; e += TPB) hA[e] = f2bf(h32[e]);
        }
        __syncthreads();

        // ---- phase A: zA = tanh(h @ W1^T + b1), K=256, N=512 (4 N-tiles/wave)
        for (int j = 0; j < 4; ++j) {
            int n0 = (wave * 4 + j) * 16;
            v8f c = {};
            #pragma unroll 4
            for (int kc = 0; kc < 8; ++kc) {
                v16bf a   = load_a_frag(hA + kc * 32, Hn);
                v16bf bfr = load_b_frag(W1b + (size_t)n0 * Hn + kc * 32, Hn);
                c = __builtin_amdgcn_wmma_f32_16x16x32_bf16(
                        false, a, false, bfr, (short)0, c, false, false);
            }
            float bias = b1g[n0 + ln];
            #pragma unroll
            for (int i = 0; i < 8; ++i)
                zA[(i + 8 * g) * Vd + n0 + ln] = f2bf(fast_tanh(c[i] + bias));
        }
        __syncthreads();
        layernorm_inplace(zA, redS, redQ, mu, rs, g1g, be1g, tid);

        // ---- phase B: zB = tanh(zA @ W2^T + b2), K=512, N=512
        for (int j = 0; j < 4; ++j) {
            int n0 = (wave * 4 + j) * 16;
            v8f c = {};
            #pragma unroll 4
            for (int kc = 0; kc < 16; ++kc) {
                v16bf a   = load_a_frag(zA + kc * 32, Vd);
                v16bf bfr = load_b_frag(W2b + (size_t)n0 * Vd + kc * 32, Vd);
                c = __builtin_amdgcn_wmma_f32_16x16x32_bf16(
                        false, a, false, bfr, (short)0, c, false, false);
            }
            float bias = b2g[n0 + ln];
            #pragma unroll
            for (int i = 0; i < 8; ++i)
                zB[(i + 8 * g) * Vd + n0 + ln] = f2bf(fast_tanh(c[i] + bias));
        }
        __syncthreads();
        layernorm_inplace(zB, redS, redQ, mu, rs, g2g, be2g, tid);

        // ---- phase C: out = zB @ W3^T + b3 (N=4096), fused einsum with dX.
        // Two N-tiles per iteration share each A fragment load.
        // N-tile hh covers columns hh*16..hh*16+15 == (h-unit hh, d=0..15).
        {
            float dx8[8];
            #pragma unroll
            for (int i = 0; i < 8; ++i) dx8[i] = dXs[(i + 8 * g) * 16 + ln];
            for (int j = 0; j < 16; ++j) {
                int hh0 = wave * 32 + 2 * j;
                int hh1 = hh0 + 1;
                v8f c0 = {}, c1 = {};
                #pragma unroll 4
                for (int kc = 0; kc < 16; ++kc) {
                    v16bf a  = load_a_frag(zB + kc * 32, Vd);
                    v16bf bf0 = load_b_frag(W3b + (size_t)hh0 * 16 * Vd + kc * 32, Vd);
                    v16bf bf1 = load_b_frag(W3b + (size_t)hh1 * 16 * Vd + kc * 32, Vd);
                    c0 = __builtin_amdgcn_wmma_f32_16x16x32_bf16(
                            false, a, false, bf0, (short)0, c0, false, false);
                    c1 = __builtin_amdgcn_wmma_f32_16x16x32_bf16(
                            false, a, false, bf1, (short)0, c1, false, false);
                }
                float b3v0 = b3g[hh0 * 16 + ln];
                float b3v1 = b3g[hh1 * 16 + ln];
                #pragma unroll
                for (int i = 0; i < 8; ++i) {
                    float v0 = (c0[i] + b3v0) * dx8[i];   // out[m][hh*16+d]*dX[m][d]
                    float v1 = (c1[i] + b3v1) * dx8[i];
                    v0 += __shfl_xor(v0, 1, 32);          // reduce over d within
                    v1 += __shfl_xor(v1, 1, 32);          // each 16-lane group
                    v0 += __shfl_xor(v0, 2, 32);
                    v1 += __shfl_xor(v1, 2, 32);
                    v0 += __shfl_xor(v0, 4, 32);
                    v1 += __shfl_xor(v1, 4, 32);
                    v0 += __shfl_xor(v0, 8, 32);
                    v1 += __shfl_xor(v1, 8, 32);
                    if (ln == 0) {
                        h32[(i + 8 * g) * Hn + hh0] += dtf * v0;  // unique writers
                        h32[(i + 8 * g) * Hn + hh1] += dtf * v1;
                    }
                }
            }
        }
        __syncthreads();
    }

    // ---- emit hT
    for (int m = 0; m < 16; ++m)
        out[(size_t)(bb + m) * Hn + tid] = h32[m * Hn + tid];
}

__global__ void f2bf_kernel(const float* __restrict__ src, uint16_t* __restrict__ dst, int n) {
    int i = blockIdx.x * blockDim.x + threadIdx.x;
    if (i < n) dst[i] = f2bf(src[i]);
}

extern "C" void kernel_launch(void* const* d_in, const int* in_sizes, int n_in,
                              void* d_out, int out_size, void* d_ws, size_t ws_size,
                              hipStream_t stream) {
    (void)in_sizes; (void)n_in; (void)out_size; (void)ws_size;
    const float* coeffs = (const float*)d_in[0];
    const float* W0  = (const float*)d_in[1];
    const float* b0  = (const float*)d_in[2];
    const float* W1  = (const float*)d_in[3];
    const float* b1  = (const float*)d_in[4];
    const float* g1  = (const float*)d_in[5];
    const float* be1 = (const float*)d_in[6];
    const float* W2  = (const float*)d_in[7];
    const float* b2  = (const float*)d_in[8];
    const float* g2  = (const float*)d_in[9];
    const float* be2 = (const float*)d_in[10];
    const float* W3  = (const float*)d_in[11];
    const float* b3  = (const float*)d_in[12];

    uint16_t* W1b = (uint16_t*)d_ws;           // 512*256
    uint16_t* W2b = W1b + 512 * 256;           // 512*512
    uint16_t* W3b = W2b + 512 * 512;           // 4096*512

    const int n1 = 512 * 256, n2 = 512 * 512, n3 = 4096 * 512;
    f2bf_kernel<<<(n1 + 255) / 256, 256, 0, stream>>>(W1, W1b, n1);
    f2bf_kernel<<<(n2 + 255) / 256, 256, 0, stream>>>(W2, W2b, n2);
    f2bf_kernel<<<(n3 + 255) / 256, 256, 0, stream>>>(W3, W3b, n3);

    ncde_kernel<<<dim3(512 / 16), dim3(TPB), 0, stream>>>(
        coeffs, W0, b0, b1, g1, be1, b2, g2, be2, b3, W1b, W2b, W3b, (float*)d_out);
}
